// TransformerEncoder_89644557403155
// MI455X (gfx1250) — compile-verified, matrix operand order checked by reading, not run on hardware
//
#include <hip/hip_runtime.h>
#include <math.h>

// ---------------- Problem constants (from reference) ----------------
#define NN   50000      // nodes (divisible by 16; 3125 row tiles)
#define EE   800000     // edges
#define INCH 128
#define HID  512        // H*C
#define HEADS 8
#define CH   64

typedef __attribute__((ext_vector_type(16))) __bf16 v16bf;
typedef __attribute__((ext_vector_type(8)))  __bf16 v8bf;
typedef __attribute__((ext_vector_type(8)))  float  v8f;

static inline int ceilDiv(int a, int b) { return (a + b - 1) / b; }

// ---------------- Elementwise / conversion kernels ----------------
__global__ void k_f32_to_bf16(const float* __restrict__ src, __bf16* __restrict__ dst, int n) {
    int i = blockIdx.x * blockDim.x + threadIdx.x;
    if (i < n) dst[i] = (__bf16)src[i];
}

__global__ void k_relu_to_bf16(const float* __restrict__ src, __bf16* __restrict__ dst, int n) {
    int i = blockIdx.x * blockDim.x + threadIdx.x;
    if (i < n) {
        float v = src[i];
        dst[i] = (__bf16)(v > 0.f ? v : 0.f);
    }
}

// Wt[m*K + k] = bf16(W[k*M + m])  (coalesced writes)
__global__ void k_wtrans(const float* __restrict__ W, __bf16* __restrict__ Wt, int K, int M) {
    int gid = blockIdx.x * blockDim.x + threadIdx.x;
    if (gid >= K * M) return;
    int m = gid / K;
    int k = gid - m * K;
    Wt[gid] = (__bf16)W[(size_t)k * M + m];
}

__global__ void k_fill_f32(float* __restrict__ p, float v, int n) {
    int i = blockIdx.x * blockDim.x + threadIdx.x;
    if (i < n) p[i] = v;
}

// ---------------- WMMA GEMM:  Y[N,Mout] = A[N,K](bf16) * Wt[Mout,K]^T(bf16) + bias ----------------
// Register-blocked: each wave computes 16 rows x 64 cols (4 accumulators). Per K-step all
// fragments (1x A, 4x B) are loaded into distinct registers BEFORE the 4 WMMAs so the
// scheduler can clause the 10 b128 loads together and overlap them with the WMMA burst
// (disjoint accumulators -> no inter-WMMA hazards, no full loadcnt drains per WMMA).
// Block = 128 threads = 4 waves splitting consecutive 16-row tiles.
// bf16 A-fragment layout (16x32): lanes 0-15 row M=lane, K in {0..7,16..23};
// lanes 16-31 row M=lane-16, K in {8..15,24..31}. B symmetric over columns.
__global__ __launch_bounds__(128)
void k_gemm_wmma(const __bf16* __restrict__ A, const __bf16* __restrict__ Wt,
                 const float* __restrict__ bias, float* __restrict__ Y,
                 int K, int Mout)
{
    const int lane    = threadIdx.x & 31;
    const int wave    = threadIdx.x >> 5;
    const int row0    = (blockIdx.x * 4 + wave) * 16;
    if (row0 >= NN) return;                 // 3125 % 4 != 0: tail guard (uniform per wave)
    const int col0    = blockIdx.y * 64;
    const int halfSel = lane >> 4;          // which 16-lane half
    const int l15     = lane & 15;
    const int koff    = halfSel * 8;        // K sub-offset per half

    const __bf16* aRow = A + (size_t)(row0 + l15) * K;
    const __bf16* bRow[4];
    #pragma unroll
    for (int j = 0; j < 4; ++j)
        bRow[j] = Wt + (size_t)(col0 + j * 16 + l15) * K;

    v8f acc[4] = {};
    for (int kk = 0; kk < K; kk += 32) {
        __builtin_prefetch(aRow + kk + 64, 0, 1);   // global_prefetch_b8 (next A chunk)

        // ---- issue ALL fragment loads for this K-step first ----
        v8bf a0 = *(const v8bf*)(aRow + kk + koff);
        v8bf a1 = *(const v8bf*)(aRow + kk + koff + 16);
        v8bf b0[4], b1[4];
        #pragma unroll
        for (int j = 0; j < 4; ++j) {
            b0[j] = *(const v8bf*)(bRow[j] + kk + koff);
            b1[j] = *(const v8bf*)(bRow[j] + kk + koff + 16);
        }

        // ---- then the 4 WMMAs (shared A fragment, disjoint accumulators) ----
        v16bf af = __builtin_shufflevector(a0, a1, 0,1,2,3,4,5,6,7,8,9,10,11,12,13,14,15);
        #pragma unroll
        for (int j = 0; j < 4; ++j) {
            v16bf bf = __builtin_shufflevector(b0[j], b1[j],
                                               0,1,2,3,4,5,6,7,8,9,10,11,12,13,14,15);
            acc[j] = __builtin_amdgcn_wmma_f32_16x16x32_bf16(
                         false, af, false, bf, (short)0, acc[j], false, false);
        }
    }

    #pragma unroll
    for (int j = 0; j < 4; ++j) {
        const float bv = bias[col0 + j * 16 + l15];
        #pragma unroll
        for (int r = 0; r < 8; ++r) {
            int row = row0 + r + halfSel * 8;   // C/D layout: vgpr r -> M=r (+8 upper half)
            Y[(size_t)row * Mout + col0 + j * 16 + l15] = acc[j][r] + bv;
        }
    }
}

// ---------------- Attention: logits + atomic segment max ----------------
__device__ inline void atomicMaxF32(float* addr, float val) {
    if (val >= 0.f) atomicMax((int*)addr, __float_as_int(val));
    else            atomicMin((unsigned int*)addr, __float_as_uint(val));
}

__global__ void k_logits(const float* __restrict__ q, const float* __restrict__ kf,
                         const long long* __restrict__ src, const long long* __restrict__ dst,
                         float* __restrict__ alpha, float* __restrict__ m,
                         int nEH, int Hh, float scale)
{
    int gid = blockIdx.x * blockDim.x + threadIdx.x;
    if (gid >= nEH) return;
    int e = gid / Hh, h = gid - e * Hh;
    int s = (int)src[e], d = (int)dst[e];
    const int F = Hh * CH;
    const float4* qp = (const float4*)(q  + (size_t)d * F + h * CH);
    const float4* kp = (const float4*)(kf + (size_t)s * F + h * CH);
    float acc = 0.f;
    #pragma unroll
    for (int i = 0; i < CH / 4; ++i) {
        float4 qa = qp[i], ka = kp[i];
        acc += qa.x * ka.x + qa.y * ka.y + qa.z * ka.z + qa.w * ka.w;
    }
    float a = acc * scale;
    alpha[gid] = a;
    atomicMaxF32(m + (size_t)d * Hh + h, a);
}

__global__ void k_fix_m(float* __restrict__ m, int n) {
    int i = blockIdx.x * blockDim.x + threadIdx.x;
    if (i < n && !isfinite(m[i])) m[i] = 0.f;
}

__global__ void k_exp_denom(float* __restrict__ alpha, const long long* __restrict__ dst,
                            const float* __restrict__ m, float* __restrict__ denom,
                            int nEH, int Hh)
{
    int gid = blockIdx.x * blockDim.x + threadIdx.x;
    if (gid >= nEH) return;
    int e = gid / Hh, h = gid - e * Hh;
    int d = (int)dst[e];
    float ea = __expf(alpha[gid] - m[(size_t)d * Hh + h]);
    alpha[gid] = ea;
    atomicAdd(denom + (size_t)d * Hh + h, ea);
}

__global__ void k_scatter(const float* __restrict__ ea, const float* __restrict__ denom,
                          const float* __restrict__ v, const long long* __restrict__ src,
                          const long long* __restrict__ dst, float* __restrict__ Y,
                          int nEH, int Hh)
{
    int gid = blockIdx.x * blockDim.x + threadIdx.x;
    if (gid >= nEH) return;
    int e = gid / Hh, h = gid - e * Hh;
    int s = (int)src[e], d = (int)dst[e];
    const int F = Hh * CH;
    float attn = ea[gid] / (denom[(size_t)d * Hh + h] + 1e-16f);
    const float4* vp = (const float4*)(v + (size_t)s * F + h * CH);
    float* yp = Y + (size_t)d * F + h * CH;
    #pragma unroll
    for (int i = 0; i < CH / 4; ++i) {
        float4 vv = vp[i];
        atomicAdd(yp + 4 * i + 0, attn * vv.x);
        atomicAdd(yp + 4 * i + 1, attn * vv.y);
        atomicAdd(yp + 4 * i + 2, attn * vv.z);
        atomicAdd(yp + 4 * i + 3, attn * vv.w);
    }
}

// ---------------- Host orchestration ----------------
extern "C" void kernel_launch(void* const* d_in, const int* in_sizes, int n_in,
                              void* d_out, int out_size, void* d_ws, size_t ws_size,
                              hipStream_t stream)
{
    (void)in_sizes; (void)n_in; (void)out_size; (void)ws_size;

    const float*     x    = (const float*)d_in[0];
    const long long* ei   = (const long long*)d_in[1];     // jnp.int64 [2,E]
    const long long* srcI = ei;
    const long long* dstI = ei + EE;

    const float *Wm[3][4], *Bv[3][4];                      // j: 0=q 1=k 2=v 3=skip
    int t = 2;
    for (int l = 0; l < 3; ++l)
        for (int j = 0; j < 4; ++j) { Wm[l][j] = (const float*)d_in[t++]; Bv[l][j] = (const float*)d_in[t++]; }

    // ---- workspace bump allocator ----
    char* ws = (char*)d_ws;
    size_t off = 0;
    auto alloc = [&](size_t bytes) -> void* {
        void* p = ws + off;
        off += (bytes + 255) & ~(size_t)255;
        return p;
    };
    __bf16* xb = (__bf16*)alloc((size_t)NN * HID * 2);     // layer input, bf16
    __bf16* wt[4];
    for (int j = 0; j < 4; ++j) wt[j] = (__bf16*)alloc((size_t)HID * HID * 2);
    float* qb    = (float*)alloc((size_t)NN * HID * 4);
    float* kb    = (float*)alloc((size_t)NN * HID * 4);
    float* vb    = (float*)alloc((size_t)NN * HID * 4);
    float* Ybuf  = (float*)alloc((size_t)NN * HID * 4);
    float* alpha = (float*)alloc((size_t)EE * HEADS * 4);
    float* mbuf  = (float*)alloc((size_t)NN * HEADS * 4);
    float* dbuf  = (float*)alloc((size_t)NN * HEADS * 4);

    const int TB = 256;
    struct { int K, F, Hh; } cfg[3] = { {INCH, HID, HEADS}, {HID, HID, HEADS}, {HID, CH, 1} };

    // layer-0 input conversion
    k_f32_to_bf16<<<ceilDiv(NN * INCH, TB), TB, 0, stream>>>(x, xb, NN * INCH);

    const int rowTiles  = NN / 16;                 // 3125
    const int rowGroups = ceilDiv(rowTiles, 4);    // 782 (tail wave guarded in-kernel)

    for (int l = 0; l < 3; ++l) {
        const int K = cfg[l].K, F = cfg[l].F, Hh = cfg[l].Hh;
        float* Y = (l == 2) ? (float*)d_out : Ybuf;

        // weights: transpose + bf16
        for (int j = 0; j < 4; ++j)
            k_wtrans<<<ceilDiv(K * F, TB), TB, 0, stream>>>(Wm[l][j], wt[j], K, F);

        // Q/K/V GEMMs + skip GEMM (skip writes Y first; attention accumulates on top)
        dim3 gg(rowGroups, F / 64);
        k_gemm_wmma<<<gg, 128, 0, stream>>>(xb, wt[0], Bv[l][0], qb, K, F);
        k_gemm_wmma<<<gg, 128, 0, stream>>>(xb, wt[1], Bv[l][1], kb, K, F);
        k_gemm_wmma<<<gg, 128, 0, stream>>>(xb, wt[2], Bv[l][2], vb, K, F);
        k_gemm_wmma<<<gg, 128, 0, stream>>>(xb, wt[3], Bv[l][3], Y,  K, F);

        // segment softmax state
        const int nNH = NN * Hh, nEH = EE * Hh;
        k_fill_f32<<<ceilDiv(nNH, TB), TB, 0, stream>>>(mbuf, -INFINITY, nNH);
        k_fill_f32<<<ceilDiv(nNH, TB), TB, 0, stream>>>(dbuf, 0.f, nNH);

        k_logits   <<<ceilDiv(nEH, TB), TB, 0, stream>>>(qb, kb, srcI, dstI, alpha, mbuf, nEH, Hh, 0.125f);
        k_fix_m    <<<ceilDiv(nNH, TB), TB, 0, stream>>>(mbuf, nNH);
        k_exp_denom<<<ceilDiv(nEH, TB), TB, 0, stream>>>(alpha, dstI, mbuf, dbuf, nEH, Hh);
        k_scatter  <<<ceilDiv(nEH, TB), TB, 0, stream>>>(alpha, dbuf, vb, srcI, dstI, Y, nEH, Hh);

        // ReLU + bf16 for next layer (layer 2 result stays f32 in d_out)
        if (l < 2)
            k_relu_to_bf16<<<ceilDiv(NN * F, TB), TB, 0, stream>>>(Y, xb, NN * F);
    }
}
